// TruthGPTModel_8555574854220
// MI455X (gfx1250) — matrix operand order repaired and checked by reading, not compiled, not run on hardware
//
#include <hip/hip_runtime.h>
#include <hip/hip_bf16.h>

// ---------------------------------------------------------------------------
// TruthGPT 12-layer encoder, distance-softmax attention, CDNA5 (gfx1250).
// GEMMs: V_WMMA_F32_16X16X32_BF16 (bf16 in, f32 accum), wave32.
// Tile staging: GLOBAL_LOAD_ASYNC_TO_LDS_B128 + s_wait_asynccnt, double-buffered.
// ---------------------------------------------------------------------------

typedef __bf16 bf16_t;
typedef __attribute__((ext_vector_type(16))) __bf16 v16bf;
typedef __attribute__((ext_vector_type(8)))  __bf16 v8bf;
typedef __attribute__((ext_vector_type(8)))  float  v8f;

#define LL     12
#define HH     768
#define NHEAD  12
#define HDIM   64
#define II     3072
#define BB     4
#define SS     1024
#define MM     (BB * SS)          // 4096 token rows
#define ATT_SCALE 0.125f          // 1/sqrt(64)

union FragBF { v16bf v; v8bf h[2]; };

__device__ __forceinline__ v8f wmma_bf16(v16bf a, v16bf b, v8f c) {
  // D = A(16x32 bf16) * B(32x16 bf16) + C(16x16 f32)
  return __builtin_amdgcn_wmma_f32_16x16x32_bf16(
      false, a, false, b, (short)0, c, false, false);
}

// Async DMA of 16 bytes per lane straight into LDS (no VGPR round trip).
// Tracked by ASYNCcnt; completion enforced with s_wait_asynccnt.
__device__ __forceinline__ void async_load_b128(unsigned int lds_off,
                                                const void* gaddr) {
  asm volatile("global_load_async_to_lds_b128 %0, %1, off"
               :: "v"(lds_off), "v"(gaddr)
               : "memory");
}

__device__ __forceinline__ void wait_async0() {
  asm volatile("s_wait_asynccnt 0x0" ::: "memory");
}

// A-operand fragment. p0 -> (row0, k0) of a 16x32 window whose rows are
// K-contiguous with `stride` elements between rows.
// Lane L<16: row L, halves K{0..7,16..23}; lane L>=16: row L-16, K{8..15,24..31}.
__device__ __forceinline__ v16bf frag_a(const bf16_t* p0, int stride, int lane) {
  const bf16_t* p = p0 + (size_t)(lane & 15) * stride + ((lane >> 4) << 3);
  FragBF f;
  f.h[0] = *(const v8bf*)(p);
  f.h[1] = *(const v8bf*)(p + 16);
  return f.v;
}

// B-operand fragment. "Columns" of B are stored as K-contiguous rows (i.e. the
// weight is pre-transposed to [N][K]). Lane L holds N=L%16, K{0..7,16..23} (L<16) / K{8..15,24..31} (L>=16).
__device__ __forceinline__ v16bf frag_b(const bf16_t* p0, int stride, int lane) {
  const bf16_t* p = p0 + (size_t)(lane & 15) * stride + ((lane >> 4) << 3);
  FragBF f;
  f.h[0] = *(const v8bf*)(p);
  f.h[1] = *(const v8bf*)(p + 16);
  return f.v;
}

// A-operand fragment sourced from f32 data (softmax probabilities in LDS),
// converted to bf16 on the fly.
__device__ __forceinline__ v16bf frag_a_f32(const float* p0, int stride, int lane) {
  const float* p = p0 + (size_t)(lane & 15) * stride + ((lane >> 4) << 3);
  v8bf lo, hi;
#pragma unroll
  for (int i = 0; i < 8; ++i) lo[i] = (bf16_t)p[i];
#pragma unroll
  for (int i = 0; i < 8; ++i) hi[i] = (bf16_t)p[16 + i];
  FragBF f; f.h[0] = lo; f.h[1] = hi;
  return f.v;
}

// ---------------------------------------------------------------------------
// GEMM: C[M,N] = A[M,K](bf16) @ Bt[N,K](bf16, pre-transposed) + bias
// epilogue: optional residual add (f32), exact GELU, f32 and/or bf16 store.
// 128x128 tile, 256 threads (8 waves), K-step 32, double-buffered async LDS.
// ---------------------------------------------------------------------------
#define MT 128
#define NT 128
#define KT 32
#define LSTR 40   // padded LDS row stride in halves (80B = 20 banks)

__global__ __launch_bounds__(256)
void gemm_bf16(const bf16_t* __restrict__ A, const bf16_t* __restrict__ Bt,
               const float* __restrict__ bias, const float* __restrict__ resid,
               float* __restrict__ outF, bf16_t* __restrict__ outB,
               int N, int K, int gelu) {
  __shared__ bf16_t As[2][MT * LSTR];
  __shared__ bf16_t Bs[2][NT * LSTR];
  const int tid = threadIdx.x, lane = tid & 31, w = tid >> 5;
  const int wm = (w >> 2) << 6;      // 0 / 64
  const int wn = (w & 3) << 5;       // 0..96
  const int mBlk = blockIdx.y * MT, nBlk = blockIdx.x * NT;

  // This thread's two fixed 16-byte staging chunks (512 chunks / 256 threads).
  const int r0 = tid >> 2,           c0 = (tid & 3) << 3;
  const int r1 = (tid + 256) >> 2,   c1 = (tid & 3) << 3;   // (tid+256)&3 == tid&3
  const bf16_t* Ag = A + (size_t)mBlk * K;
  const bf16_t* Bg = Bt + (size_t)nBlk * K;

  v8f acc[4][2];
#pragma unroll
  for (int mf = 0; mf < 4; ++mf)
#pragma unroll
    for (int nf = 0; nf < 2; ++nf) acc[mf][nf] = (v8f){};

  // Prologue: async-stage tile 0 into buffer 0.
  async_load_b128((unsigned int)(uintptr_t)&As[0][r0 * LSTR + c0],
                  &Ag[(size_t)r0 * K + c0]);
  async_load_b128((unsigned int)(uintptr_t)&Bs[0][r0 * LSTR + c0],
                  &Bg[(size_t)r0 * K + c0]);
  async_load_b128((unsigned int)(uintptr_t)&As[0][r1 * LSTR + c1],
                  &Ag[(size_t)r1 * K + c1]);
  async_load_b128((unsigned int)(uintptr_t)&Bs[0][r1 * LSTR + c1],
                  &Bg[(size_t)r1 * K + c1]);
  wait_async0();
  __syncthreads();

  int buf = 0;
  for (int k0 = 0; k0 < K; k0 += KT) {
    // Kick off the DMA for the next tile into the other buffer.
    if (k0 + KT < K) {
      const int nb = buf ^ 1, kn = k0 + KT;
      async_load_b128((unsigned int)(uintptr_t)&As[nb][r0 * LSTR + c0],
                      &Ag[(size_t)r0 * K + kn + c0]);
      async_load_b128((unsigned int)(uintptr_t)&Bs[nb][r0 * LSTR + c0],
                      &Bg[(size_t)r0 * K + kn + c0]);
      async_load_b128((unsigned int)(uintptr_t)&As[nb][r1 * LSTR + c1],
                      &Ag[(size_t)r1 * K + kn + c1]);
      async_load_b128((unsigned int)(uintptr_t)&Bs[nb][r1 * LSTR + c1],
                      &Bg[(size_t)r1 * K + kn + c1]);
    }

    // Compute current tile while the DMA runs.
    v16bf af[4], bfr[2];
#pragma unroll
    for (int mf = 0; mf < 4; ++mf)
      af[mf] = frag_a(&As[buf][(wm + mf * 16) * LSTR], LSTR, lane);
#pragma unroll
    for (int nf = 0; nf < 2; ++nf)
      bfr[nf] = frag_b(&Bs[buf][(wn + nf * 16) * LSTR], LSTR, lane);
#pragma unroll
    for (int mf = 0; mf < 4; ++mf)
#pragma unroll
      for (int nf = 0; nf < 2; ++nf)
        acc[mf][nf] = wmma_bf16(af[mf], bfr[nf], acc[mf][nf]);

    wait_async0();        // my async writes to buf^1 have landed in LDS
    __syncthreads();      // everyone done reading buf / writing buf^1
    buf ^= 1;
  }

  // Epilogue. C layout: VGPR r -> M = r + 8*(lane>=16), N = lane%16.
#pragma unroll
  for (int mf = 0; mf < 4; ++mf)
#pragma unroll
    for (int nf = 0; nf < 2; ++nf) {
      int mBase = mBlk + wm + mf * 16 + ((lane >> 4) << 3);
      int n = nBlk + wn + nf * 16 + (lane & 15);
      float bi = bias[n];
#pragma unroll
      for (int r = 0; r < 8; ++r) {
        int m = mBase + r;
        float val = acc[mf][nf][r] + bi;
        if (resid) val += resid[(size_t)m * N + n];
        if (gelu)  val = 0.5f * val * (1.f + erff(val * 0.70710678118f));
        if (outF)  outF[(size_t)m * N + n] = val;
        if (outB)  outB[(size_t)m * N + n] = (bf16_t)val;
      }
    }
}

// ---------------------------------------------------------------------------
// Distance-softmax attention. One WG per (head, 32-row q-tile), 512 thr = 16
// waves. Phase 1: QK^T via WMMA (16 waves x 64 key-cols), score into LDS.
// Phase 2: row softmax. Phase 3: P@V split-K across waves, ds_add_f32 combine.
// ---------------------------------------------------------------------------
#define QT 32

__global__ __launch_bounds__(512)
void attn_dist(const bf16_t* __restrict__ Qh, const bf16_t* __restrict__ Kh,
               const bf16_t* __restrict__ Vt, const float* __restrict__ qq,
               const float* __restrict__ kk, const float* __restrict__ lam,
               bf16_t* __restrict__ AO, int layer) {
  __shared__ float sc[QT][SS];     // 128 KB score tile
  __shared__ float ao[QT][HDIM];   // 8 KB split-K combine buffer
  const int tid = threadIdx.x, lane = tid & 31, w = tid >> 5;
  const int bh = blockIdx.y, q0 = blockIdx.x * QT;
  const float lamv = lam[layer];

  for (int i = tid; i < QT * HDIM; i += 512) ((float*)ao)[i] = 0.f;

  // ---- Phase 1: scores = -lam * sqrt(max(qq+kk-2*QK^T,0)+eps) * scale ----
  const bf16_t* Qb = Qh + ((size_t)bh * SS + q0) * HDIM;
  v16bf qf[2][2];
#pragma unroll
  for (int mf = 0; mf < 2; ++mf)
#pragma unroll
    for (int ks = 0; ks < 2; ++ks)
      qf[mf][ks] = frag_a(Qb + mf * 16 * HDIM + ks * 32, HDIM, lane);

  const int nW = w * 64;                       // this wave's 64 key columns
  const float* qqr = qq + (size_t)bh * SS + q0;
  const float* kkr = kk + (size_t)bh * SS;
#pragma unroll
  for (int nf = 0; nf < 4; ++nf) {
    v8f a0 = (v8f){}, a1 = (v8f){};
    const bf16_t* Kb = Kh + ((size_t)bh * SS + nW + nf * 16) * HDIM;
#pragma unroll
    for (int ks = 0; ks < 2; ++ks) {
      v16bf kf = frag_b(Kb + ks * 32, HDIM, lane);
      a0 = wmma_bf16(qf[0][ks], kf, a0);
      a1 = wmma_bf16(qf[1][ks], kf, a1);
    }
    int n = nW + nf * 16 + (lane & 15);
    float kkv = kkr[n];
#pragma unroll
    for (int r = 0; r < 8; ++r) {
      int m0 = r + ((lane >> 4) << 3);
      float d2 = qqr[m0] + kkv - 2.f * a0[r];
      sc[m0][n] = -lamv * sqrtf(fmaxf(d2, 0.f) + 1e-8f) * ATT_SCALE;
      int m1 = m0 + 16;
      d2 = qqr[m1] + kkv - 2.f * a1[r];
      sc[m1][n] = -lamv * sqrtf(fmaxf(d2, 0.f) + 1e-8f) * ATT_SCALE;
    }
  }
  __syncthreads();

  // ---- Phase 2: softmax (wave w owns rows 2w, 2w+1) ----
#pragma unroll
  for (int rr = 0; rr < 2; ++rr) {
    int r = w * 2 + rr;
    float mx = -3.4e38f;
    for (int c = lane; c < SS; c += 32) mx = fmaxf(mx, sc[r][c]);
#pragma unroll
    for (int o = 16; o; o >>= 1) mx = fmaxf(mx, __shfl_xor(mx, o, 32));
    float sum = 0.f;
    for (int c = lane; c < SS; c += 32) {
      float e = __expf(sc[r][c] - mx);
      sc[r][c] = e;
      sum += e;
    }
#pragma unroll
    for (int o = 16; o; o >>= 1) sum += __shfl_xor(sum, o, 32);
    float inv = 1.f / sum;
    for (int c = lane; c < SS; c += 32) sc[r][c] *= inv;
  }
  __syncthreads();

  // ---- Phase 3: P @ V, split-K (wave w owns keys [64w, 64w+64)) ----
  const int kB = w * 64;
  v8f acc[2][4];
#pragma unroll
  for (int mf = 0; mf < 2; ++mf)
#pragma unroll
    for (int nf = 0; nf < 4; ++nf) acc[mf][nf] = (v8f){};
#pragma unroll
  for (int ks = 0; ks < 2; ++ks) {
    v16bf p0 = frag_a_f32(&sc[0][kB + ks * 32], SS, lane);
    v16bf p1 = frag_a_f32(&sc[16][kB + ks * 32], SS, lane);
#pragma unroll
    for (int nf = 0; nf < 4; ++nf) {
      const bf16_t* Vb = Vt + ((size_t)bh * HDIM + nf * 16) * SS + kB + ks * 32;
      v16bf vf = frag_b(Vb, SS, lane);
      acc[0][nf] = wmma_bf16(p0, vf, acc[0][nf]);
      acc[1][nf] = wmma_bf16(p1, vf, acc[1][nf]);
    }
  }
#pragma unroll
  for (int mf = 0; mf < 2; ++mf)
#pragma unroll
    for (int nf = 0; nf < 4; ++nf) {
      int m = mf * 16 + ((lane >> 4) << 3);
      int d = nf * 16 + (lane & 15);
#pragma unroll
      for (int r = 0; r < 8; ++r) atomicAdd(&ao[m + r][d], acc[mf][nf][r]);
    }
  __syncthreads();

  const int b = bh / NHEAD, head = bh % NHEAD;
  for (int i = tid; i < QT * HDIM; i += 512) {
    int m = i >> 6, d = i & 63;
    AO[((size_t)b * SS + q0 + m) * HH + head * HDIM + d] = (bf16_t)ao[m][d];
  }
}

// ---------------------------------------------------------------------------
// LayerNorm (768 cols, one row per 256-thread block), bf16 output for GEMM A.
// ---------------------------------------------------------------------------
__device__ __forceinline__ float block_sum_256(float v) {
  __shared__ float red[8];
  const int lane = threadIdx.x & 31, w = threadIdx.x >> 5;
  __syncthreads();
#pragma unroll
  for (int o = 16; o; o >>= 1) v += __shfl_xor(v, o, 32);
  if (lane == 0) red[w] = v;
  __syncthreads();
  float t = (lane < 8) ? red[lane] : 0.f;
#pragma unroll
  for (int o = 16; o; o >>= 1) t += __shfl_xor(t, o, 32);
  return t;
}

__global__ __launch_bounds__(256)
void layernorm_bf16(const float* __restrict__ h, const float* __restrict__ g,
                    const float* __restrict__ b, bf16_t* __restrict__ out) {
  const int row = blockIdx.x;
  const float* x = h + (size_t)row * HH;
  float v[3];
#pragma unroll
  for (int i = 0; i < 3; ++i) v[i] = x[threadIdx.x + i * 256];
  float mean = block_sum_256(v[0] + v[1] + v[2]) * (1.f / HH);
  float s2 = 0.f;
#pragma unroll
  for (int i = 0; i < 3; ++i) { float d = v[i] - mean; s2 += d * d; }
  float inv = rsqrtf(block_sum_256(s2) * (1.f / HH) + 1e-5f);
#pragma unroll
  for (int i = 0; i < 3; ++i) {
    int c = threadIdx.x + i * 256;
    out[(size_t)row * HH + c] = (bf16_t)((v[i] - mean) * inv * g[c] + b[c]);
  }
}

// ---------------------------------------------------------------------------
// Head rearrange kernels.
// ---------------------------------------------------------------------------
__global__ __launch_bounds__(64)
void head_split_qk(const float* __restrict__ X, bf16_t* __restrict__ Out,
                   float* __restrict__ ss) {
  const int idx = blockIdx.x;
  const int head = idx % NHEAD;
  const int s = (idx / NHEAD) % SS;
  const int b = idx / (NHEAD * SS);
  const int d = threadIdx.x;
  float v = X[((size_t)b * SS + s) * HH + head * HDIM + d];
  const int bh = b * NHEAD + head;
  Out[((size_t)bh * SS + s) * HDIM + d] = (bf16_t)v;
  float p = v * v;
#pragma unroll
  for (int o = 16; o; o >>= 1) p += __shfl_xor(p, o, 32);
  __shared__ float r2[2];
  if ((threadIdx.x & 31) == 0) r2[threadIdx.x >> 5] = p;
  __syncthreads();
  if (threadIdx.x == 0) ss[(size_t)bh * SS + s] = r2[0] + r2[1];
}

__global__ __launch_bounds__(256)
void head_split_v(const float* __restrict__ X, bf16_t* __restrict__ Vt) {
  size_t i = (size_t)blockIdx.x * 256 + threadIdx.x;   // over [B*NH][HD][S]
  int s = (int)(i % SS);
  int d = (int)((i / SS) % HDIM);
  int bh = (int)(i / ((size_t)SS * HDIM));
  int b = bh / NHEAD, head = bh % NHEAD;
  Vt[i] = (bf16_t)X[((size_t)b * SS + s) * HH + head * HDIM + d];
}

// Weight convert + transpose: W[K,N] f32 -> Wt[N,K] bf16.
__global__ __launch_bounds__(256)
void conv_transpose_w(const float* __restrict__ W, bf16_t* __restrict__ Wt,
                      int K, int N) {
  size_t i = (size_t)blockIdx.x * 256 + threadIdx.x;
  if (i >= (size_t)K * N) return;
  int n = (int)(i % N);
  int k = (int)(i / N);
  Wt[(size_t)n * K + k] = (bf16_t)W[i];
}

// ---------------------------------------------------------------------------
// Host orchestration.
// ---------------------------------------------------------------------------
extern "C" void kernel_launch(void* const* d_in, const int* in_sizes, int n_in,
                              void* d_out, int out_size, void* d_ws, size_t ws_size,
                              hipStream_t stream) {
  (void)in_sizes; (void)n_in; (void)out_size; (void)ws_size;
  const float* hs  = (const float*)d_in[0];
  const float* Wq  = (const float*)d_in[1];
  const float* bq  = (const float*)d_in[2];
  const float* Wk  = (const float*)d_in[3];
  const float* bk  = (const float*)d_in[4];
  const float* Wv  = (const float*)d_in[5];
  const float* bv  = (const float*)d_in[6];
  const float* Wo  = (const float*)d_in[7];
  const float* bo  = (const float*)d_in[8];
  const float* g1  = (const float*)d_in[9];
  const float* be1 = (const float*)d_in[10];
  const float* g2  = (const float*)d_in[11];
  const float* be2 = (const float*)d_in[12];
  const float* W1  = (const float*)d_in[13];
  const float* b1  = (const float*)d_in[14];
  const float* W2  = (const float*)d_in[15];
  const float* b2  = (const float*)d_in[16];
  const float* lam = (const float*)d_in[17];

  char* wp = (char*)d_ws;
  auto alloc = [&](size_t bytes) -> void* {
    void* r = (void*)wp;
    wp += (bytes + 255) & ~(size_t)255;
    return r;
  };
  bf16_t* WqT = (bf16_t*)alloc((size_t)LL * HH * HH * 2);
  bf16_t* WkT = (bf16_t*)alloc((size_t)LL * HH * HH * 2);
  bf16_t* WvT = (bf16_t*)alloc((size_t)LL * HH * HH * 2);
  bf16_t* WoT = (bf16_t*)alloc((size_t)LL * HH * HH * 2);
  bf16_t* W1T = (bf16_t*)alloc((size_t)LL * HH * II * 2);
  bf16_t* W2T = (bf16_t*)alloc((size_t)LL * HH * II * 2);
  bf16_t* xb  = (bf16_t*)alloc((size_t)MM * HH * 2);
  bf16_t* yb  = (bf16_t*)alloc((size_t)MM * HH * 2);
  float*  qf32 = (float*)alloc((size_t)MM * HH * 4);
  float*  kf32 = (float*)alloc((size_t)MM * HH * 4);
  float*  vf32 = (float*)alloc((size_t)MM * HH * 4);
  bf16_t* Qhh = (bf16_t*)alloc((size_t)MM * HH * 2);
  bf16_t* Khh = (bf16_t*)alloc((size_t)MM * HH * 2);
  bf16_t* Vtt = (bf16_t*)alloc((size_t)MM * HH * 2);
  bf16_t* AO  = (bf16_t*)alloc((size_t)MM * HH * 2);
  float*  qqA = (float*)alloc((size_t)BB * NHEAD * SS * 4);
  float*  kkA = (float*)alloc((size_t)BB * NHEAD * SS * 4);
  bf16_t* ub  = (bf16_t*)alloc((size_t)MM * II * 2);

  float* h = (float*)d_out;                       // residual stream lives in d_out
  hipMemcpyAsync(h, hs, (size_t)MM * HH * 4, hipMemcpyDeviceToDevice, stream);

  // Weight pre-pass: fp32 -> bf16, transposed to [N][K].
  const int gSq  = (HH * HH + 255) / 256;
  const int gBig = (HH * II + 255) / 256;
  for (int l = 0; l < LL; ++l) {
    size_t o2 = (size_t)l * HH * HH, o3 = (size_t)l * HH * II;
    conv_transpose_w<<<gSq, 256, 0, stream>>>(Wq + o2, WqT + o2, HH, HH);
    conv_transpose_w<<<gSq, 256, 0, stream>>>(Wk + o2, WkT + o2, HH, HH);
    conv_transpose_w<<<gSq, 256, 0, stream>>>(Wv + o2, WvT + o2, HH, HH);
    conv_transpose_w<<<gSq, 256, 0, stream>>>(Wo + o2, WoT + o2, HH, HH);
    conv_transpose_w<<<gBig, 256, 0, stream>>>(W1 + o3, W1T + o3, HH, II);
    conv_transpose_w<<<gBig, 256, 0, stream>>>(W2 + o3, W2T + o3, II, HH);
  }

  const dim3 gProj(HH / NT, MM / MT);   // (6, 32)
  const dim3 gFfn1(II / NT, MM / MT);   // (24, 32)
  const dim3 gAttn(SS / QT, BB * NHEAD);

  for (int l = 0; l < LL; ++l) {
    const size_t oW = (size_t)l * HH * HH, oB = (size_t)l * HH;
    const size_t oW1 = (size_t)l * HH * II, oI = (size_t)l * II;

    layernorm_bf16<<<MM, 256, 0, stream>>>(h, g1 + oB, be1 + oB, xb);

    gemm_bf16<<<gProj, 256, 0, stream>>>(xb, WqT + oW, bq + oB, nullptr,
                                         qf32, nullptr, HH, HH, 0);
    gemm_bf16<<<gProj, 256, 0, stream>>>(xb, WkT + oW, bk + oB, nullptr,
                                         kf32, nullptr, HH, HH, 0);
    gemm_bf16<<<gProj, 256, 0, stream>>>(xb, WvT + oW, bv + oB, nullptr,
                                         vf32, nullptr, HH, HH, 0);

    head_split_qk<<<BB * SS * NHEAD, 64, 0, stream>>>(qf32, Qhh, qqA);
    head_split_qk<<<BB * SS * NHEAD, 64, 0, stream>>>(kf32, Khh, kkA);
    head_split_v<<<(BB * NHEAD * HDIM * SS) / 256, 256, 0, stream>>>(vf32, Vtt);

    attn_dist<<<gAttn, 512, 0, stream>>>(Qhh, Khh, Vtt, qqA, kkA, lam, AO, l);

    gemm_bf16<<<gProj, 256, 0, stream>>>(AO, WoT + oW, bo + oB, h,
                                         h, nullptr, HH, HH, 0);

    layernorm_bf16<<<MM, 256, 0, stream>>>(h, g2 + oB, be2 + oB, yb);

    gemm_bf16<<<gFfn1, 256, 0, stream>>>(yb, W1T + oW1, b1 + oI, nullptr,
                                         nullptr, ub, II, HH, 1);
    gemm_bf16<<<gProj, 256, 0, stream>>>(ub, W2T + oW1, b2 + oB, h,
                                         h, nullptr, HH, II, 0);
  }
}
